// HGATLayer_84310208021181
// MI455X (gfx1250) — matrix operand order repaired and checked by reading, not compiled, not run on hardware
//
#include <hip/hip_runtime.h>
#include <hip/hip_bf16.h>

typedef __attribute__((ext_vector_type(16))) __bf16 v16bf;
typedef __attribute__((ext_vector_type(8)))  float  v8f;
typedef __attribute__((ext_vector_type(4)))  int    v4i;

union fragu { v16bf bf; v4i i2[2]; };          // one WMMA 16-elem bf16 fragment
union bpu   { __bf16 h[2]; unsigned int u; };  // packed bf16 pair

#define N_NODES 8192
#define M_EDGES 4096
#define FIN     256
#define FOUT    128
#define LALPHA  0.2f

__device__ __forceinline__ float leaky(float v) { return v > 0.f ? v : LALPHA * v; }

// inverse of A-fragment K mapping: K -> (half, e)
__device__ __forceinline__ int a_half(int K) { return (K >> 4) & 1; }
__device__ __forceinline__ int a_elem(int K) { return K & 15; }

// ---------------------------------------------------------------------------
// K0: pack a [K=rows][128] f32 matrix into B-fragment bf16 layout
//     pk[((k/32)*128 + d)*32 + k%32]
// ---------------------------------------------------------------------------
__global__ void k_wpack(const float* __restrict__ w, __bf16* __restrict__ wpk)
{
    int t = blockIdx.x * 256 + threadIdx.x;    // FIN*FOUT total
    int k = t >> 7, d = t & 127;
    wpk[((size_t)(k >> 5) * FOUT + d) * 32 + (k & 31)] = (__bf16)w[t];
}

// ---------------------------------------------------------------------------
// K1: xt = x @ weight, x4 = x @ weight2. Block = 8 waves, 16 rows, waves split D.
// A staged into LDS pre-packed in fragment order; B read as 2x b128 from wpk.
// ---------------------------------------------------------------------------
__global__ void k_xform(const float* __restrict__ x,
                        const __bf16* __restrict__ w1pk,
                        const __bf16* __restrict__ w2pk,
                        float* __restrict__ xt,
                        float* __restrict__ x4)
{
    __shared__ __bf16 lds_a[32][16];
    const int tid  = threadIdx.x;
    const int wave = tid >> 5, lane = tid & 31;
    const int half = lane >> 4, ln = lane & 15;
    const int i0 = blockIdx.x * 16;
    const int d0 = wave * 16;
    v8f acc_t = {}; v8f acc_4 = {};
    for (int kk = 0; kk < FIN; kk += 32) {
        {   // coalesced read of x tile [16 m x 32 K], write packed A frags
            int idx = tid * 2;
            int r = idx >> 5, c = idx & 31;               // m=r, K=c (even)
            const float2 v = *(const float2*)(x + (size_t)(i0 + r) * FIN + kk + c);
            bpu p; p.h[0] = (__bf16)v.x; p.h[1] = (__bf16)v.y;
            *(unsigned int*)&lds_a[r + a_half(c) * 16][a_elem(c)] = p.u;
        }
        __syncthreads();
        fragu a, bt, b4;
        a.i2[0] = *(const v4i*)&lds_a[lane][0];
        a.i2[1] = *(const v4i*)&lds_a[lane][8];
        size_t base = ((size_t)(kk >> 5) * FOUT + d0 + ln) * 32 + half * 16;
        bt.i2[0] = *(const v4i*)(w1pk + base);
        bt.i2[1] = *(const v4i*)(w1pk + base + 8);
        b4.i2[0] = *(const v4i*)(w2pk + base);
        b4.i2[1] = *(const v4i*)(w2pk + base + 8);
        acc_t = __builtin_amdgcn_wmma_f32_16x16x32_bf16(false, a.bf, false, bt.bf, (short)0, acc_t, false, false);
        acc_4 = __builtin_amdgcn_wmma_f32_16x16x32_bf16(false, a.bf, false, b4.bf, (short)0, acc_4, false, false);
        __syncthreads();
    }
#pragma unroll
    for (int r = 0; r < 8; ++r) {
        int row = i0 + r + half * 8;
        int col = d0 + ln;
        xt[(size_t)row * FOUT + col] = acc_t[r];
        x4[(size_t)row * FOUT + col] = acc_4[r];
    }
}

// ---------------------------------------------------------------------------
// K2: per-node GEMVs: s1[j] = leaky(wc.a[:D] + x4[j].a[D:]),  f[j] = x4[j].a2[:D]
// ---------------------------------------------------------------------------
__global__ void k_scores(const float* __restrict__ x4,
                         const float* __restrict__ wc,
                         const float* __restrict__ a,
                         const float* __restrict__ a2,
                         float* __restrict__ s1, float* __restrict__ fv)
{
    const int wave = threadIdx.x >> 5, lane = threadIdx.x & 31;
    const int j = blockIdx.x * 8 + wave;
    const float* row = x4 + (size_t)j * FOUT;
    float p1 = 0.f, p2 = 0.f, p3 = 0.f;
#pragma unroll
    for (int t = 0; t < 4; ++t) {
        int d = lane * 4 + t;
        float xv = row[d];
        p1 += xv * a[FOUT + d];
        p2 += wc[d] * a[d];
        p3 += xv * a2[d];
    }
#pragma unroll
    for (int off = 16; off > 0; off >>= 1) {
        p1 += __shfl_down(p1, off);
        p2 += __shfl_down(p2, off);
        p3 += __shfl_down(p3, off);
    }
    if (lane == 0) { s1[j] = leaky(p2 + p1); fv[j] = p3; }
}

// ---------------------------------------------------------------------------
// K3: out[0] = max(v1), out[1] = max(v2)   (global softmax shifts)
// ---------------------------------------------------------------------------
__global__ void k_rmax2(const float* __restrict__ v1, int n1,
                        const float* __restrict__ v2, int n2,
                        float* __restrict__ out)
{
    __shared__ float red[256];
    const int tid = threadIdx.x;
    float m1 = -3.4e38f;
    for (int i = tid; i < n1; i += 256) m1 = fmaxf(m1, v1[i]);
    red[tid] = m1; __syncthreads();
    for (int s = 128; s > 0; s >>= 1) {
        if (tid < s) red[tid] = fmaxf(red[tid], red[tid + s]);
        __syncthreads();
    }
    if (tid == 0) out[0] = red[0];
    __syncthreads();
    float m2 = -3.4e38f;
    for (int i = tid; i < n2; i += 256) m2 = fmaxf(m2, v2[i]);
    red[tid] = m2; __syncthreads();
    for (int s = 128; s > 0; s >>= 1) {
        if (tid < s) red[tid] = fmaxf(red[tid], red[tid + s]);
        __syncthreads();
    }
    if (tid == 0) out[1] = red[0];
}

// ---------------------------------------------------------------------------
// K4: e1[j] = exp(s1[j]-gmax);  ypk = B-fragment-packed bf16(e1[j]*xt[j,k]).
// Block handles a 32-row j slab; LDS transpose for coalesced in/out.
// ---------------------------------------------------------------------------
__global__ void k_pack_y(const float* __restrict__ s1, const float* __restrict__ gmax,
                         const float* __restrict__ xt,
                         float* __restrict__ e1, __bf16* __restrict__ ypk)
{
    __shared__ float xs[32][FOUT + 1];
    __shared__ float e1_s[32];
    const int t = threadIdx.x;
    const int jb = blockIdx.x * 32;
#pragma unroll
    for (int q = 0; q < 4; ++q) {
        int row = (t >> 5) + q * 8;
        int c4  = (t & 31) * 4;
        const float4 v = *(const float4*)(xt + (size_t)(jb + row) * FOUT + c4);
        xs[row][c4] = v.x; xs[row][c4 + 1] = v.y; xs[row][c4 + 2] = v.z; xs[row][c4 + 3] = v.w;
    }
    if (t < 32) {
        float ev = __expf(s1[jb + t] - gmax[0]);
        e1_s[t] = ev; e1[jb + t] = ev;
    }
    __syncthreads();
    const int k = t >> 1, j2 = (t & 1) * 16;
    fragu f;
#pragma unroll
    for (int q = 0; q < 16; ++q) f.bf[q] = (__bf16)(e1_s[j2 + q] * xs[j2 + q][k]);
    v4i* dst = (v4i*)(ypk + ((size_t)blockIdx.x * FOUT + k) * 32 + j2);
    dst[0] = f.i2[0]; dst[1] = f.i2[1];
}

// ---------------------------------------------------------------------------
// K5 (big GEMM 1): edge[i,:] = (sum_j adj[j,i]*e1[j]*xt[j,:]) / (sum_j adj[j,i]*e1[j])
// A = adj^T (exact 0/1 bf16), staged into packed-fragment LDS; B = ypk (2x b128).
// Denominator in f32 from the A fragments (wave 0 only). adj read exactly once.
// ---------------------------------------------------------------------------
__global__ void k_edge(const float* __restrict__ adj,
                       const float* __restrict__ e1,
                       const __bf16* __restrict__ ypk,
                       float* __restrict__ edge_f, __bf16* __restrict__ epk)
{
    __shared__ __bf16 lds_a[32][16];
    __shared__ float e1_s[32];
    __shared__ float denom_s[16];
    const int tid  = threadIdx.x;
    const int wave = tid >> 5, lane = tid & 31;
    const int half = lane >> 4, ln = lane & 15;
    const int i0 = blockIdx.x * 16;
    const int d0 = wave * 16;
    v8f acc = {};
    float dpart = 0.f;
    for (int jb = 0; jb < N_NODES; jb += 32) {
        {   // coalesced read adj[jb+r][i0+c..c+1]; element (m=i=c, K=j=r)
            int idx = tid * 2;
            int r = idx >> 4, c = idx & 15;
            const float* p = adj + (size_t)(jb + r) * M_EDGES + i0 + c;
            const float2 v = *(const float2*)p;
            int h = a_half(r), e0 = a_elem(r);
            lds_a[c + h * 16][e0]     = (__bf16)v.x;
            lds_a[c + 1 + h * 16][e0] = (__bf16)v.y;
            if (jb + 256 + r < N_NODES)
                __builtin_prefetch(adj + (size_t)(jb + 256 + r) * M_EDGES + i0 + c, 0, 1);
        }
        if (tid < 32) e1_s[tid] = e1[jb + tid];
        __syncthreads();
        fragu a, b;
        a.i2[0] = *(const v4i*)&lds_a[lane][0];
        a.i2[1] = *(const v4i*)&lds_a[lane][8];
        size_t base = ((size_t)(jb >> 5) * FOUT + d0 + ln) * 32 + half * 16;
        b.i2[0] = *(const v4i*)(ypk + base);
        b.i2[1] = *(const v4i*)(ypk + base + 8);
        acc = __builtin_amdgcn_wmma_f32_16x16x32_bf16(false, a.bf, false, b.bf, (short)0, acc, false, false);
        if (wave == 0) {   // f32 denominator: each (i,j) exactly once across lanes
#pragma unroll
            for (int e = 0; e < 16; ++e) {
                int ka = e + (half ? 16 : 0);
                dpart += (float)a.bf[e] * e1_s[ka];
            }
        }
        __syncthreads();
    }
    if (wave == 0) {
        float other = __shfl_down(dpart, 16);
        if (lane < 16) denom_s[lane] = dpart + other;
    }
    __syncthreads();
#pragma unroll
    for (int r = 0; r < 8; ++r) {
        int mi = r + half * 8;
        int i = i0 + mi, d = d0 + ln;
        float dv = fmaxf(denom_s[mi], 1e-30f);
        float v = acc[r] / dv;
        edge_f[(size_t)i * FOUT + d] = v;
        epk[((size_t)(i >> 5) * FOUT + d) * 32 + (i & 31)] = (__bf16)v;  // B-frag pack
    }
}

// ---------------------------------------------------------------------------
// K6: u[k] = sum_d W3[k,d]*a2[D+d]   (collapses edge4: e2 = edge @ u)
// ---------------------------------------------------------------------------
__global__ void k_u(const float* __restrict__ w3, const float* __restrict__ a2,
                    float* __restrict__ u)
{
    const int k = threadIdx.x;
    float s = 0.f;
    for (int d = 0; d < FOUT; ++d) s += w3[(size_t)k * FOUT + d] * a2[FOUT + d];
    u[k] = s;
}

// ---------------------------------------------------------------------------
// K7: e2[i] = edge[i,:] . u   (one wave per row)
// ---------------------------------------------------------------------------
__global__ void k_e2(const float* __restrict__ edge_f, const float* __restrict__ u,
                     float* __restrict__ e2)
{
    const int wave = threadIdx.x >> 5, lane = threadIdx.x & 31;
    const int i = blockIdx.x * 8 + wave;
    const float* row = edge_f + (size_t)i * FOUT;
    float p = 0.f;
#pragma unroll
    for (int t = 0; t < 4; ++t) { int d = lane * 4 + t; p += row[d] * u[d]; }
#pragma unroll
    for (int off = 16; off > 0; off >>= 1) p += __shfl_down(p, off);
    if (lane == 0) e2[i] = p;
}

// ---------------------------------------------------------------------------
// K8: S = leaky(max(e2) + fmax)   (global shift; leaky is monotone)
// ---------------------------------------------------------------------------
__global__ void k_rmaxS(const float* __restrict__ v, int n,
                        const float* __restrict__ fmaxp, float* __restrict__ Sout)
{
    __shared__ float red[256];
    const int tid = threadIdx.x;
    float m = -3.4e38f;
    for (int i = tid; i < n; i += 256) m = fmaxf(m, v[i]);
    red[tid] = m; __syncthreads();
    for (int s = 128; s > 0; s >>= 1) {
        if (tid < s) red[tid] = fmaxf(red[tid], red[tid + s]);
        __syncthreads();
    }
    if (tid == 0) Sout[0] = leaky(red[0] + fmaxp[0]);
}

// ---------------------------------------------------------------------------
// K9 (big GEMM 2): node[j,:] = (sum_i w[j,i]*edge[i,:]) / sum_i w[j,i],
//   w[j,i] = adj[j,i]*exp(leaky(e2[i]+f[j]) - S);  out = ELU(node).
// w computed cooperatively straight into packed-fragment LDS (one b32 store per
// pair via v_cvt_pk); denominators in f32; B = epk (2x b128).
// ---------------------------------------------------------------------------
__global__ void k_node(const float* __restrict__ adj,
                       const float* __restrict__ e2,
                       const float* __restrict__ fv,
                       const float* __restrict__ Sp,
                       const __bf16* __restrict__ epk,
                       float* __restrict__ out)
{
    __shared__ __bf16 lds_a[32][16];
    __shared__ float e2_s[32];
    __shared__ float denom_s[16];
    const int tid  = threadIdx.x;
    const int wave = tid >> 5, lane = tid & 31;
    const int half = lane >> 4, ln = lane & 15;
    const int j0 = blockIdx.x * 16;
    const int d0 = wave * 16;
    const int rj = tid >> 4;          // cooperative row (j) for this thread
    const int c2 = (tid & 15) * 2;    // its two i-columns (even)
    const float S = Sp[0];
    const float frow = fv[j0 + rj];
    if (tid < 16) denom_s[tid] = 0.f;
    float dAcc = 0.f;
    v8f acc = {};
    __syncthreads();
    const int wh = a_half(c2), we = a_elem(c2);   // pair lands in one lane slot
    for (int ib = 0; ib < M_EDGES; ib += 32) {
        if (tid < 32) e2_s[tid] = e2[ib + tid];
        __syncthreads();
        {
            const float* p = adj + (size_t)(j0 + rj) * M_EDGES + ib + c2;
            const float2 av = *(const float2*)p;
            if (ib + 256 < M_EDGES) __builtin_prefetch(p + 256, 0, 1);
            float w0 = av.x * __expf(leaky(e2_s[c2]     + frow) - S);
            float w1 = av.y * __expf(leaky(e2_s[c2 + 1] + frow) - S);
            bpu pk; pk.h[0] = (__bf16)w0; pk.h[1] = (__bf16)w1;
            *(unsigned int*)&lds_a[rj + wh * 16][we] = pk.u;  // A[m=j][K=i]
            dAcc += w0 + w1;
        }
        __syncthreads();
        fragu a, b;
        a.i2[0] = *(const v4i*)&lds_a[lane][0];
        a.i2[1] = *(const v4i*)&lds_a[lane][8];
        size_t base = ((size_t)(ib >> 5) * FOUT + d0 + ln) * 32 + half * 16;
        b.i2[0] = *(const v4i*)(epk + base);
        b.i2[1] = *(const v4i*)(epk + base + 8);
        acc = __builtin_amdgcn_wmma_f32_16x16x32_bf16(false, a.bf, false, b.bf, (short)0, acc, false, false);
        __syncthreads();
    }
    atomicAdd(&denom_s[rj], dAcc);
    __syncthreads();
#pragma unroll
    for (int r = 0; r < 8; ++r) {
        int mj = r + half * 8;
        float dv = fmaxf(denom_s[mj], 1e-30f);
        float v = acc[r] / dv;
        v = v > 0.f ? v : (__expf(v) - 1.f);          // ELU (alpha=1)
        out[(size_t)(j0 + mj) * FOUT + d0 + ln] = v;
    }
}

// ---------------------------------------------------------------------------
extern "C" void kernel_launch(void* const* d_in, const int* in_sizes, int n_in,
                              void* d_out, int out_size, void* d_ws, size_t ws_size,
                              hipStream_t stream)
{
    (void)in_sizes; (void)n_in; (void)out_size; (void)ws_size;
    const float* x   = (const float*)d_in[0];
    const float* adj = (const float*)d_in[1];
    const float* w1  = (const float*)d_in[2];   // weight
    const float* w2  = (const float*)d_in[3];   // weight2
    const float* w3  = (const float*)d_in[4];   // weight3
    const float* wc  = (const float*)d_in[5];   // word_context [128]
    const float* av  = (const float*)d_in[6];   // a  [256]
    const float* a2  = (const float*)d_in[7];   // a2 [256]

    // workspace layout (packed-bf16 regions 16B aligned)
    float*  x4    = (float*)d_ws;                       // N*FOUT
    float*  xt    = x4    + (size_t)N_NODES * FOUT;     // N*FOUT
    float*  edgef = xt    + (size_t)N_NODES * FOUT;     // M*FOUT
    float*  s1    = edgef + (size_t)M_EDGES * FOUT;     // N
    float*  fvv   = s1 + N_NODES;                       // N
    float*  e1    = fvv + N_NODES;                      // N
    float*  uvec  = e1 + N_NODES;                       // FOUT
    float*  e2v   = uvec + FOUT;                        // M
    float*  scal  = e2v + M_EDGES;                      // [0]=gmax1 [1]=fmax [2]=S (pad 16)
    __bf16* ypk   = (__bf16*)(scal + 16);               // N*FOUT packed
    __bf16* epk   = ypk  + (size_t)N_NODES * FOUT;      // M*FOUT packed
    __bf16* w1pk  = epk  + (size_t)M_EDGES * FOUT;      // FIN*FOUT packed
    __bf16* w2pk  = w1pk + (size_t)FIN * FOUT;          // FIN*FOUT packed

    k_wpack <<<(FIN * FOUT) / 256, 256, 0, stream>>>(w1, w1pk);
    k_wpack <<<(FIN * FOUT) / 256, 256, 0, stream>>>(w2, w2pk);
    k_xform <<<N_NODES / 16, 256, 0, stream>>>(x, w1pk, w2pk, xt, x4);
    k_scores<<<N_NODES / 8,  256, 0, stream>>>(x4, wc, av, a2, s1, fvv);
    k_rmax2 <<<1, 256, 0, stream>>>(s1, N_NODES, fvv, N_NODES, scal);
    k_pack_y<<<N_NODES / 32, 256, 0, stream>>>(s1, scal, xt, e1, ypk);
    k_edge  <<<M_EDGES / 16, 256, 0, stream>>>(adj, e1, ypk, edgef, epk);
    k_u     <<<1, FOUT, 0, stream>>>(w3, a2, uvec);
    k_e2    <<<M_EDGES / 8, 256, 0, stream>>>(edgef, uvec, e2v);
    k_rmaxS <<<1, 256, 0, stream>>>(e2v, M_EDGES, scal + 1, scal + 2);
    k_node  <<<N_NODES / 16, 256, 0, stream>>>(adj, e2v, fvv, scal + 2, epk, (float*)d_out);
}
